// FakeFusedExperts_56014963474857
// MI455X (gfx1250) — matrix operand order repaired and checked by reading, not compiled, not run on hardware
//
#include <hip/hip_runtime.h>
#include <hip/hip_bf16.h>

// Problem constants (match reference)
#define E_  32
#define H_  1024
#define F_  512
#define T_  1024
#define K_  8
#define CAP (T_ * K_)   // per-expert worst-case routed entries (duplicates allowed)

typedef __bf16 bf16_t;
typedef __attribute__((ext_vector_type(16))) __bf16 v16bf;
typedef __attribute__((ext_vector_type(8)))  float  v8f;

__device__ __forceinline__ v8f wmma_bf16(v16bf a, v16bf b, v8f c) {
  // D = A(16x32 bf16) * B(32x16 bf16) + C(16x16 f32)
  return __builtin_amdgcn_wmma_f32_16x16x32_bf16(
      /*neg_a=*/false, a, /*neg_b=*/false, b,
      /*c_mod=*/(short)0, c, /*reuse_a=*/false, /*reuse_b=*/false);
}

// A-fragment (16x32, 16-bit) gathered from LDS row-major [16][ld].
// Layout: lanes 0-15 -> M=0..15, elements hold K=0..15 contiguous;
// lanes 16-31 hold K=16..31 (matches load_b_global).
__device__ __forceinline__ v16bf load_a_lds(const bf16_t* base, int ld, int k0, int lane) {
  const int g = lane >> 4;
  const int m = lane & 15;
  const bf16_t* row = base + m * ld + k0 + g * 16;
  v16bf a;
#pragma unroll
  for (int r = 0; r < 8; ++r) {
    const int kk = 2 * r;
    a[2 * r]     = row[kk];
    a[2 * r + 1] = row[kk + 1];
  }
  return a;
}

// B-fragment (32x16, 16-bit): B[k][n] = W[n][k], W row-major fp32 with row stride ld.
// Lanes 0-15 hold K=0..15 (col N = lane), lanes 16-31 hold K=16..31; VGPR r: K = g*16 + {2r, 2r+1}.
__device__ __forceinline__ v16bf load_b_global(const float* __restrict__ W, int ld,
                                               int n0, int k0, int lane) {
  const int g = lane >> 4;
  const int c = lane & 15;
  const float* col = W + (size_t)(n0 + c) * ld + (k0 + g * 16);
  v16bf b;
#pragma unroll
  for (int r = 0; r < 8; ++r) {
    const float2 f = *(const float2*)(col + 2 * r);
    b[2 * r]     = (bf16_t)f.x;
    b[2 * r + 1] = (bf16_t)f.y;
  }
  return b;
}

__device__ __forceinline__ void atomic_add_f32(float* p, float v) {
  __hip_atomic_fetch_add(p, v, __ATOMIC_RELAXED, __HIP_MEMORY_SCOPE_AGENT);
}

// ---------------- kernel 1: zero output + expert counters ----------------
__global__ void moe_zero_kernel(float* __restrict__ out, int n, int* __restrict__ counts) {
  const int i = blockIdx.x * blockDim.x + threadIdx.x;
  if (i < n) out[i] = 0.0f;
  if (i < E_) counts[i] = 0;
}

// ---------------- kernel 2: build per-expert routing lists ----------------
__global__ void moe_route_kernel(const int* __restrict__ topk_idx,
                                 const float* __restrict__ topk_w,
                                 int* __restrict__ counts,
                                 int* __restrict__ tlist,
                                 float* __restrict__ wlist) {
  const int i = blockIdx.x * blockDim.x + threadIdx.x;  // over T*K
  if (i >= T_ * K_) return;
  const int t = i / K_;
  const int e = topk_idx[i];
  const int p = atomicAdd(&counts[e], 1);
  tlist[(size_t)e * CAP + p] = t;
  wlist[(size_t)e * CAP + p] = topk_w[i];
}

// ---------------- kernel 3: grouped-GEMM fused expert FFN ----------------
// grid = (CAP/16, E), block = 256 threads = 8 wave32
__global__ __launch_bounds__(256)
void moe_ffn_kernel(const float* __restrict__ hidden,
                    const float* __restrict__ gate_up,   // (E, 2F, H)
                    const float* __restrict__ down,      // (E, H, F)
                    const int* __restrict__ counts,
                    const int* __restrict__ tlist,
                    const float* __restrict__ wlist,
                    float* __restrict__ out) {
  const int e    = blockIdx.y;
  const int tile = blockIdx.x;
  const int n_e  = counts[e];
  if (tile * 16 >= n_e) return;

  __shared__ bf16_t Xs[16][H_];    // 32 KB: gathered token tile, bf16
  __shared__ bf16_t Acts[16][F_];  // 16 KB: silu(gate)*up tile, bf16
  __shared__ int    toks_s[16];
  __shared__ float  wts_s[16];

  const int tid  = threadIdx.x;
  const int wave = tid >> 5;
  const int lane = tid & 31;

  if (tid < 16) {
    const int p = tile * 16 + tid;
    if (p < n_e) {
      toks_s[tid] = tlist[(size_t)e * CAP + p];
      wts_s[tid]  = wlist[(size_t)e * CAP + p];
    } else {
      toks_s[tid] = 0;      // padded row: weight 0 contributes nothing
      wts_s[tid]  = 0.0f;
    }
  }
  __syncthreads();

  // Cooperative coalesced stage of X tile (fp32 -> bf16)
  for (int i = tid; i < 16 * H_; i += 256) {
    const int r = i >> 10;          // H_ == 1024
    const int c = i & (H_ - 1);
    Xs[r][c] = (bf16_t)hidden[(size_t)toks_s[r] * H_ + c];
  }
  __syncthreads();

  // ---- phase 1: gu = X @ Wgu^T, per-wave 64 columns of gate and of up ----
  const float* Wgu = gate_up + (size_t)e * (2 * F_) * H_;
  v8f accg[4] = {};
  v8f accu[4] = {};
  for (int k0 = 0; k0 < H_; k0 += 32) {
    const v16bf a = load_a_lds(&Xs[0][0], H_, k0, lane);
#pragma unroll
    for (int ft = 0; ft < 4; ++ft) {
      const int f0 = wave * 64 + ft * 16;
      const v16bf bg = load_b_global(Wgu, H_, f0,       k0, lane);
      const v16bf bu = load_b_global(Wgu, H_, F_ + f0,  k0, lane);
      accg[ft] = wmma_bf16(a, bg, accg[ft]);
      accu[ft] = wmma_bf16(a, bu, accu[ft]);
    }
  }

  // ---- SiLU(gate) * up -> Acts (bf16) ----
  {
    const int g  = lane >> 4;
    const int cc = lane & 15;
#pragma unroll
    for (int ft = 0; ft < 4; ++ft) {
      const int f0 = wave * 64 + ft * 16;
#pragma unroll
      for (int r = 0; r < 8; ++r) {
        const int M = r + 8 * g;     // C/D layout: VGPR r, lane half g
        const float gv = accg[ft][r];
        const float uv = accu[ft][r];
        const float s  = gv / (1.0f + __expf(-gv));
        Acts[M][f0 + cc] = (bf16_t)(s * uv);
      }
    }
  }
  __syncthreads();

  // ---- phase 2: out_tile = Acts @ Wd^T, per-wave 128 columns of H ----
  const float* Wd = down + (size_t)e * H_ * F_;
  v8f acc[8] = {};
  for (int k0 = 0; k0 < F_; k0 += 32) {
    const v16bf a = load_a_lds(&Acts[0][0], F_, k0, lane);
#pragma unroll
    for (int ht = 0; ht < 8; ++ht) {
      const int h0 = wave * 128 + ht * 16;
      const v16bf b = load_b_global(Wd, F_, h0, k0, lane);
      acc[ht] = wmma_bf16(a, b, acc[ht]);
    }
  }

  // ---- weighted scatter-accumulate into d_out ----
  {
    const int g  = lane >> 4;
    const int cc = lane & 15;
#pragma unroll
    for (int ht = 0; ht < 8; ++ht) {
      const int h0 = wave * 128 + ht * 16;
#pragma unroll
      for (int r = 0; r < 8; ++r) {
        const int M = r + 8 * g;
        const float v = acc[ht][r] * wts_s[M];
        atomic_add_f32(&out[(size_t)toks_s[M] * H_ + h0 + cc], v);
      }
    }
  }
}

extern "C" void kernel_launch(void* const* d_in, const int* in_sizes, int n_in,
                              void* d_out, int out_size, void* d_ws, size_t ws_size,
                              hipStream_t stream) {
  const float* hidden   = (const float*)d_in[0];   // (T, H)
  const int*   topk_idx = (const int*)  d_in[1];   // (T, K)
  const float* topk_w   = (const float*)d_in[2];   // (T, K)
  const float* gup      = (const float*)d_in[3];   // (E, 2F, H)
  const float* dwn      = (const float*)d_in[4];   // (E, H, F)
  float*       out      = (float*)d_out;           // (T, H)

  // workspace layout
  char* ws = (char*)d_ws;
  int*   counts = (int*)ws;                                          // E ints
  int*   tlist  = (int*)(ws + 256);                                  // E*CAP ints (1 MB)
  float* wlist  = (float*)(ws + 256 + sizeof(int) * (size_t)E_ * CAP); // E*CAP floats (1 MB)

  moe_zero_kernel<<<dim3((T_ * H_ + 255) / 256), 256, 0, stream>>>(out, T_ * H_, counts);
  moe_route_kernel<<<dim3((T_ * K_ + 255) / 256), 256, 0, stream>>>(topk_idx, topk_w,
                                                                    counts, tlist, wlist);
  moe_ffn_kernel<<<dim3(CAP / 16, E_), 256, 0, stream>>>(hidden, gup, dwn,
                                                         counts, tlist, wlist, out);
}
